// ColorHistograms_41351945126516
// MI455X (gfx1250) — matrix operand order repaired and checked by reading, not compiled, not run on hardware
//
#include <hip/hip_runtime.h>
#include <hip/hip_fp16.h>

typedef __attribute__((ext_vector_type(16))) _Float16 v16h;
typedef __attribute__((ext_vector_type(8)))  _Float16 v8h;
typedef __attribute__((ext_vector_type(8)))  float    v8f;

#define B_DIM 16
#define T_DIM 512
#define NBINS 512
#define HW_PX 1296
#define LWIN 101
#define PADW 50
#define KPAD 128
#define ODIM 128

union V16 { v16h v; struct { v8h lo; v8h hi; } p; };

struct __align__(4) Px3 { int x, y, z; };   // one pixel = 3 dwords -> b96 load

// ---------------------------------------------------------------------------
// Pass 1: per-frame 512-bin color histogram + L2 normalize, write f16 rows.
// One block per frame (8192 blocks). HBM-bound: reads 127 MB of int32 frames
// with one contiguous b96 load per pixel.
// ---------------------------------------------------------------------------
__global__ __launch_bounds__(256)
void hist_kernel(const int* __restrict__ frames, _Float16* __restrict__ hist)
{
    __shared__ int   histo[NBINS];
    __shared__ float red[256];
    const int f   = blockIdx.x;
    const int tid = threadIdx.x;

    histo[tid]       = 0;
    histo[tid + 256] = 0;
    __syncthreads();

    const Px3* px = reinterpret_cast<const Px3*>(frames + (long long)f * (HW_PX * 3));
    for (int p = tid; p < HW_PX; p += 256) {
        Px3 v = px[p];
        int r = (v.x >> 5) & 7;
        int g = (v.y >> 5) & 7;
        int b = (v.z >> 5) & 7;
        atomicAdd(&histo[(r << 6) | (g << 3) | b], 1);
    }
    __syncthreads();

    float h0 = (float)histo[tid];
    float h1 = (float)histo[tid + 256];
    red[tid] = h0 * h0 + h1 * h1;
    __syncthreads();
    for (int s = 128; s > 0; s >>= 1) {
        if (tid < s) red[tid] += red[tid + s];
        __syncthreads();
    }
    const float inv = 1.0f / fmaxf(sqrtf(red[0]), 1e-12f);

    _Float16* row = hist + (long long)f * NBINS;
    row[tid]       = (_Float16)(h0 * inv);
    row[tid + 256] = (_Float16)(h1 * inv);
}

// ---------------------------------------------------------------------------
// Pass 1b: W (128 x 101 f32) -> Wp (128 x 128 f16, zero-padded K).
// ---------------------------------------------------------------------------
__global__ __launch_bounds__(256)
void prepw_kernel(const float* __restrict__ W, _Float16* __restrict__ Wp)
{
    const int idx = blockIdx.x * 256 + threadIdx.x;   // 0 .. 16383
    const int o = idx >> 7;
    const int l = idx & 127;
    const float v = (l < LWIN) ? W[o * LWIN + l] : 0.0f;
    Wp[idx] = (_Float16)v;
}

// ---------------------------------------------------------------------------
// Pass 1c: zero the padded band buffer (2 MB) so pad lanes / out-of-band
// entries are 0 (ws is poisoned by the harness).
// ---------------------------------------------------------------------------
__global__ __launch_bounds__(256)
void zero_band_kernel(uint4* __restrict__ p)
{
    p[blockIdx.x * 256 + threadIdx.x] = uint4{0u, 0u, 0u, 0u};
}

// ---------------------------------------------------------------------------
// Pass 2: banded x . x^T via v_wmma_f32_16x16x32_f16. One wave per
// (batch, row-tile i, col-tile j) with j in [i-4, i+4]  (band half-width 50).
// Distinct (i,j) write disjoint banded[t][l] elements -> race-free scatter.
// A/B fragments double-buffered so loads overlap the WMMA chain.
// ---------------------------------------------------------------------------
__global__ __launch_bounds__(32)
void band_kernel(const _Float16* __restrict__ hist, _Float16* __restrict__ banded)
{
    const int b  = blockIdx.x >> 5;
    const int it = blockIdx.x & 31;                  // row tile 0..31
    const int j  = it - 4 + (int)blockIdx.y;         // col tile
    if (j < 0 || j > 31) return;

    const int lane  = threadIdx.x;
    const int half_ = lane >> 4;                     // 0 or 1
    const int lm    = lane & 15;

    const _Float16* xb   = hist + (long long)b * T_DIM * NBINS;
    const _Float16* arow = xb + (long long)(it * 16 + lm) * NBINS;
    const _Float16* brow = xb + (long long)(j * 16 + lm) * NBINS;

    // A (16x32 f16): lane holds M=lm; K = k0+16*half_..+16 (contiguous)
    // B (32x16 f16): lane holds N=lm; K = k0+16*half_..+16 (contiguous row of X)
    V16 a;  v16h bm;
    a.p.lo = *(const v8h*)(arow + 16 * half_);
    a.p.hi = *(const v8h*)(arow + 8 + 16 * half_);
    bm     = *(const v16h*)(brow + 16 * half_);

    v8f c = {};
#pragma unroll
    for (int k0 = 0; k0 < NBINS; k0 += 32) {
        V16 an; v16h bn;
        if (k0 + 32 < NBINS) {                        // prefetch next chunk
            an.p.lo = *(const v8h*)(arow + k0 + 32 + 16 * half_);
            an.p.hi = *(const v8h*)(arow + k0 + 40 + 16 * half_);
            bn      = *(const v16h*)(brow + k0 + 32 + 16 * half_);
        }
        c = __builtin_amdgcn_wmma_f32_16x16x32_f16(
                false, a.v, false, bm, (short)0, c, false, false);
        if (k0 + 32 < NBINS) { a = an; bm = bn; }
    }

    // C/D layout: VGPR v -> M = v + 8*half_, N = lm
#pragma unroll
    for (int v = 0; v < 8; ++v) {
        const int t = it * 16 + v + 8 * half_;
        const int s = j * 16 + lm;
        const int d = s - t;
        if (d >= -PADW && d <= PADW)
            banded[(long long)(b * T_DIM + t) * KPAD + (d + PADW)] = (_Float16)c[v];
    }
}

// ---------------------------------------------------------------------------
// Pass 3: out = relu(banded @ Wp^T + bias). M=8192, N=128, K=128 (padded).
// One wave per 16x16 output tile; 4 WMMAs along K.
// ---------------------------------------------------------------------------
__global__ __launch_bounds__(32)
void out_kernel(const _Float16* __restrict__ banded,
                const _Float16* __restrict__ Wp,
                const float* __restrict__ bias,
                float* __restrict__ out)
{
    const int mt    = blockIdx.x >> 3;        // row tile over B*T (0..511)
    const int cn    = blockIdx.x & 7;         // col tile (0..7)
    const int lane  = threadIdx.x;
    const int half_ = lane >> 4;
    const int lm    = lane & 15;

    const _Float16* arow = banded + (long long)(mt * 16 + lm) * KPAD;
    const _Float16* brow = Wp + (long long)(cn * 16 + lm) * KPAD;

    v8f c = {};
#pragma unroll
    for (int k0 = 0; k0 < KPAD; k0 += 32) {
        V16 a;
        a.p.lo = *(const v8h*)(arow + k0 + 16 * half_);
        a.p.hi = *(const v8h*)(arow + k0 + 8 + 16 * half_);
        v16h bm = *(const v16h*)(brow + k0 + 16 * half_);
        c = __builtin_amdgcn_wmma_f32_16x16x32_f16(
                false, a.v, false, bm, (short)0, c, false, false);
    }

    const int   o  = cn * 16 + lm;
    const float bo = bias[o];
#pragma unroll
    for (int v = 0; v < 8; ++v) {
        const int row = mt * 16 + v + 8 * half_;
        out[(long long)row * ODIM + o] = fmaxf(c[v] + bo, 0.0f);
    }
}

// ---------------------------------------------------------------------------
extern "C" void kernel_launch(void* const* d_in, const int* in_sizes, int n_in,
                              void* d_out, int out_size, void* d_ws, size_t ws_size,
                              hipStream_t stream)
{
    const int*   frames = (const int*)d_in[0];
    const float* W      = (const float*)d_in[1];
    const float* bias   = (const float*)d_in[2];
    float*       out    = (float*)d_out;

    // workspace layout (all 256B-aligned offsets):
    //   hist   : B*T*NBINS f16  = 8 MB
    //   banded : B*T*KPAD  f16  = 2 MB
    //   Wp     : ODIM*KPAD f16  = 32 KB
    char* ws = (char*)d_ws;
    const size_t hist_bytes = (size_t)B_DIM * T_DIM * NBINS * sizeof(_Float16);
    const size_t band_bytes = (size_t)B_DIM * T_DIM * KPAD * sizeof(_Float16);
    _Float16* hist   = (_Float16*)ws;
    _Float16* banded = (_Float16*)(ws + hist_bytes);
    _Float16* Wp     = (_Float16*)(ws + hist_bytes + band_bytes);

    hist_kernel<<<B_DIM * T_DIM, 256, 0, stream>>>(frames, hist);
    prepw_kernel<<<(ODIM * KPAD) / 256, 256, 0, stream>>>(W, Wp);
    zero_band_kernel<<<(int)(band_bytes / (256 * 16)), 256, 0, stream>>>((uint4*)banded);
    band_kernel<<<dim3(B_DIM * 32, 9), 32, 0, stream>>>(hist, banded);
    out_kernel<<<(B_DIM * T_DIM / 16) * 8, 32, 0, stream>>>(banded, Wp, bias, out);
}
